// RelativeDotProductAttention_24257975287923
// MI455X (gfx1250) — compile-verified, matrix operand order checked by reading, not run on hardware
//
#include <hip/hip_runtime.h>

#define MAX_REL 513
#define V_SZ    (2 * MAX_REL + 1)   // 1027
#define B_SZ    4
#define H_SZ    8
#define L_SZ    1024
#define D_SZ    64
#define BH      (B_SZ * H_SZ)       // 32
#define M_TILE  16
#define NXBLK   (L_SZ / M_TILE)     // 64 row-blocks
#define S_PITCH  1028               // 1024 + 4: 1028%64==4  -> conflict-free A-frag reads
#define W2_W     1056               // 33*32 logical width (GEMM tiles)
#define W2_PITCH 1060               // 1056 + 4: 1060%64==36 -> conflict-free A-frag reads
#define QE_TILES 65                 // ceil(1027/16)

typedef __attribute__((ext_vector_type(16))) __bf16 v16bf;
typedef __attribute__((ext_vector_type(8)))  float  v8f;

// K-index inside a 16x32 bf16 A fragment (CDNA5 ISA 7.12.2 layout), mirrored for B.
// halves h=0..15 per lane; hi = lane>>4 selects the +8 K offset group.
__device__ __forceinline__ int frag_k(int h, int hi) {
  int vg = h >> 1;
  int base = (vg < 4) ? (vg * 2) : (16 + (vg - 4) * 2);
  return base + (h & 1) + hi * 8;
}

__device__ __forceinline__ float wave_max(float v) {
  for (int off = 16; off > 0; off >>= 1) v = fmaxf(v, __shfl_xor(v, off, 32));
  return v;
}
__device__ __forceinline__ float wave_sum(float v) {
  for (int off = 16; off > 0; off >>= 1) v += __shfl_xor(v, off, 32);
  return v;
}

__global__ __launch_bounds__(128)
void relattn_kernel(const float* __restrict__ q, const float* __restrict__ k,
                    const float* __restrict__ v, const float* __restrict__ bias,
                    const float* __restrict__ ek, const float* __restrict__ ev,
                    float* __restrict__ out, float* __restrict__ wout)
{
  __shared__ float S[M_TILE][S_PITCH];    // ~64KB: logits -> exp -> normalized weights
  __shared__ float W2[M_TILE][W2_PITCH];  // ~66KB: first QE (q . emb_k^T), then rel-bin weights

  const int tid  = threadIdx.x;
  const int lane = tid & 31;
  const int wave = tid >> 5;
  const int ln   = lane & 15;  // M index (A frags / C rows) or N index (B frags / C cols)
  const int hi   = lane >> 4;

  const int xblk = blockIdx.x % NXBLK;
  const int bh   = blockIdx.x / NXBLK;
  const int x0   = xblk * M_TILE;

  const float* qb    = q    + (size_t)bh * L_SZ * D_SZ;
  const float* kb    = k    + (size_t)bh * L_SZ * D_SZ;
  const float* vb    = v    + (size_t)bh * L_SZ * D_SZ;
  const float* biasb = bias + (size_t)bh * L_SZ * L_SZ;

  // ---- Q A-fragments: 16x32 bf16, two K-chunks covering D=64 (same in all waves)
  v16bf aq[2];
  {
    const float* qrow = qb + (size_t)(x0 + ln) * D_SZ;
    for (int c = 0; c < 2; ++c)
      for (int h = 0; h < 16; ++h)
        aq[c][h] = (__bf16)qrow[c * 32 + frag_k(h, hi)];
  }

  // ---- QE[m][r] = q[m,:] . emb_keys[r,:]  (GEMM into W2 buffer)
  for (int t = wave; t < QE_TILES; t += 4) {
    int r0 = t * 16;
    v8f c = {};
    for (int ch = 0; ch < 2; ++ch) {
      int row = r0 + ln;
      row = row < V_SZ ? row : V_SZ - 1;          // tail clamp (results unused)
      const float* er = ek + (size_t)row * D_SZ + ch * 32;
      v16bf b;
      for (int h = 0; h < 16; ++h) b[h] = (__bf16)er[frag_k(h, hi)];
      c = __builtin_amdgcn_wmma_f32_16x16x32_bf16(false, aq[ch], false, b,
                                                  (short)0, c, false, false);
    }
    for (int i = 0; i < 8; ++i) W2[i + 8 * hi][r0 + ln] = c[i];
  }
  __syncthreads();

  // ---- logits tile panel: S = Q K^T + gather(QE, idx) + bias
  for (int t = wave; t < (L_SZ / 16); t += 4) {
    int y0 = t * 16;
    // prefetch next owned k tile one iteration ahead (global_prefetch_b8)
    if (t + 4 < (L_SZ / 16))
      __builtin_prefetch(kb + (size_t)((t + 4) * 16 + ln) * D_SZ + hi * 32, 0, 1);
    v8f c = {};
    for (int ch = 0; ch < 2; ++ch) {
      const float* kr = kb + (size_t)(y0 + ln) * D_SZ + ch * 32;
      v16bf b;
      for (int h = 0; h < 16; ++h) b[h] = (__bf16)kr[frag_k(h, hi)];
      c = __builtin_amdgcn_wmma_f32_16x16x32_bf16(false, aq[ch], false, b,
                                                  (short)0, c, false, false);
    }
    for (int i = 0; i < 8; ++i) {
      int m  = i + 8 * hi;
      int gx = x0 + m;
      int y  = y0 + ln;
      int d  = y - gx;
      d = d < -MAX_REL ? -MAX_REL : (d > MAX_REL ? MAX_REL : d);
      S[m][y] = c[i] + W2[m][d + MAX_REL] + biasb[(size_t)gx * L_SZ + y];
    }
  }
  __syncthreads();

  // ---- row softmax + stream weights to global (coalesced)
  float* woutb = wout + (size_t)bh * L_SZ * L_SZ;
  for (int m = wave * 4; m < wave * 4 + 4; ++m) {
    float mx = -3.4e38f;
    for (int y = lane; y < L_SZ; y += 32) mx = fmaxf(mx, S[m][y]);
    mx = wave_max(mx);
    float sum = 0.f;
    for (int y = lane; y < L_SZ; y += 32) {
      float e = __expf(S[m][y] - mx);
      S[m][y] = e;
      sum += e;
    }
    sum = wave_sum(sum);
    float inv = 1.f / sum;
    for (int y = lane; y < L_SZ; y += 32) {
      float w = S[m][y] * inv;
      S[m][y] = w;
      woutb[(size_t)(x0 + m) * L_SZ + y] = w;
    }
  }
  __syncthreads();

  // ---- fold weights into relative bins: W2[m][r] = sum_{y: idx(x,y)=r} w[m][y]
  for (int p = tid; p < M_TILE * W2_PITCH; p += 128)
    W2[p / W2_PITCH][p % W2_PITCH] = 0.f;
  __syncthreads();
  for (int p = tid; p < M_TILE * L_SZ; p += 128) {
    int m = p >> 10, y = p & (L_SZ - 1);
    int d = y - (x0 + m);
    d = d < -MAX_REL ? -MAX_REL : (d > MAX_REL ? MAX_REL : d);
    atomicAdd(&W2[m][d + MAX_REL], S[m][y]);   // ds_add_f32
  }
  __syncthreads();

  // ---- out = W @ V + W2 @ emb_values ; each wave owns one 16-wide d-tile
  const int d0 = wave * 16;
  v8f acc = {};
  for (int t = 0; t < L_SZ / 32; ++t) {
    int y0 = t * 32;
    if (t + 1 < L_SZ / 32)      // prefetch next v panel rows
      __builtin_prefetch(vb + (size_t)(y0 + 32 + lane) * D_SZ + d0, 0, 1);
    v16bf a, b;
    for (int h = 0; h < 16; ++h) {
      int kk = frag_k(h, hi);
      a[h] = (__bf16)S[ln][y0 + kk];
      b[h] = (__bf16)vb[(size_t)(y0 + kk) * D_SZ + d0 + ln];
    }
    acc = __builtin_amdgcn_wmma_f32_16x16x32_bf16(false, a, false, b,
                                                  (short)0, acc, false, false);
  }
  for (int t = 0; t < W2_W / 32; ++t) {
    int r0 = t * 32;
    v16bf a, b;
    for (int h = 0; h < 16; ++h) {
      int kk  = frag_k(h, hi);
      a[h] = (__bf16)W2[ln][r0 + kk];
      int row = r0 + kk; row = row < V_SZ ? row : V_SZ - 1; // pad rows have A==0
      b[h] = (__bf16)ev[(size_t)row * D_SZ + d0 + ln];
    }
    acc = __builtin_amdgcn_wmma_f32_16x16x32_bf16(false, a, false, b,
                                                  (short)0, acc, false, false);
  }
  float* ob = out + (size_t)bh * L_SZ * D_SZ;
  for (int i = 0; i < 8; ++i)
    ob[(size_t)(x0 + i + 8 * hi) * D_SZ + d0 + ln] = acc[i];
}

extern "C" void kernel_launch(void* const* d_in, const int* in_sizes, int n_in,
                              void* d_out, int out_size, void* d_ws, size_t ws_size,
                              hipStream_t stream) {
  const float* q    = (const float*)d_in[0];
  const float* k    = (const float*)d_in[1];
  const float* v    = (const float*)d_in[2];
  const float* bias = (const float*)d_in[3];
  const float* ek   = (const float*)d_in[4];
  const float* ev   = (const float*)d_in[5];
  float* out  = (float*)d_out;                          // (B,H,L,D) first
  float* wout = out + (size_t)BH * L_SZ * D_SZ;         // then (B,H,L,L) weights

  dim3 grid(BH * NXBLK);   // 2048 workgroups: (b*h) x 64 row-blocks
  dim3 block(128);         // 4 x wave32
  hipLaunchKernelGGL(relattn_kernel, grid, block, 0, stream,
                     q, k, v, bias, ek, ev, out, wout);
}
